// Transformer_46067819217383
// MI455X (gfx1250) — compile-verified, matrix operand order checked by reading, not run on hardware
//
#include <hip/hip_runtime.h>
#include <hip/hip_bf16.h>
#include <math.h>

// ---------------------------------------------------------------------------
// MI455X / gfx1250 fused transformer block.
// One workgroup (8 wave32s) per batch item; all GEMMs via
// v_wmma_f32_16x16x32_bf16; activations resident in ~295KB LDS.
// ---------------------------------------------------------------------------

typedef __bf16 bf16_t;
typedef __attribute__((ext_vector_type(16))) __bf16 v16bf;
typedef __attribute__((ext_vector_type(8)))  float  v8f;

#define Bn 1024
#define Nn 128
#define Dd 256
#define Ff 128
#define INV_SCALE 0.08838834764831843f   // 1/sqrt(128)

// LDS layout (bytes)
#define OFF_KVB 0        // bf16[128][256]  kv = concat(seq, seq_t)
#define OFF_KB  65536    // bf16[128][256]  K_sa -> sa_out -> ca_k
#define OFF_VTB 131072   // bf16[256][128]  V_sa^T ; later bf16[128][256] V_ca
#define OFF_AOB 196608   // bf16[128][256]  attn-out -> h1 -> K_ca
#define OFF_SCR 262144   // bf16[8][16][128] per-wave staging (Q block / attn)
#define OFF_MSK 294912   // float[128]
#define OFF_F   295424   // float scratch region (1672 floats)
#define SMEM_BYTES 302112

// WMMA 16x16x32 bf16 fragment index map (ISA 7.12.2, 16-bit A 16x32):
// lane half h, element j -> K offset
__device__ __forceinline__ int kmap(int j, int h) {
  return (j & 7) + ((j >> 3) << 4) + (h << 3);
}

// Load A-fragment from row-major [rows x ld] bf16 at (row0, k0).
// Also used for B-fragments in "Bt" form (row index = output column).
__device__ __forceinline__ v16bf load_frag(const bf16_t* p, int ld, int row0,
                                           int k0, int lane) {
  int r = lane & 15, h = lane >> 4;
  const bf16_t* q = p + (size_t)(row0 + r) * ld + k0;
  v16bf a;
#pragma unroll
  for (int j = 0; j < 16; ++j) a[j] = q[kmap(j, h)];
  return a;
}

__device__ __forceinline__ v8f vzero() {
  v8f c;
#pragma unroll
  for (int i = 0; i < 8; ++i) c[i] = 0.0f;
  return c;
}

__device__ __forceinline__ v8f wmma_bf(v16bf a, v16bf b, v8f c) {
  return __builtin_amdgcn_wmma_f32_16x16x32_bf16(false, a, false, b, (short)0,
                                                 c, false, false);
}

// Store C (16x16 f32) as bf16 row-major: lane=col n, VGPR i = row i+8*half.
__device__ __forceinline__ void store_c(bf16_t* p, int ld, int row0, int col0,
                                        v8f c, int lane) {
  int n = lane & 15, h = lane >> 4;
#pragma unroll
  for (int i = 0; i < 8; ++i)
    p[(size_t)(row0 + i + 8 * h) * ld + col0 + n] = (bf16_t)c[i];
}

// Store C transposed (for building V^T).
__device__ __forceinline__ void store_cT(bf16_t* p, int ld, int row0, int col0,
                                         v8f c, int lane) {
  int n = lane & 15, h = lane >> 4;
#pragma unroll
  for (int i = 0; i < 8; ++i)
    p[(size_t)(col0 + n) * ld + row0 + i + 8 * h] = (bf16_t)c[i];
}

// Weight prep: W[K][N] f32 -> Wt[N][K] bf16 (so B-fragments load "Bt" form).
__global__ void wprep(const float* __restrict__ w, bf16_t* __restrict__ wt,
                      int K, int N) {
  int i = blockIdx.x * 256 + threadIdx.x;
  if (i < K * N) {
    int k = i / N, n = i - k * N;
    wt[(size_t)n * K + k] = (bf16_t)w[i];
  }
}

__global__ __launch_bounds__(256) void transformer_fused(
    const float* __restrict__ src, const float* __restrict__ src_t,
    const float* __restrict__ seq, const float* __restrict__ seq_t,
    const unsigned char* __restrict__ mask,
    const bf16_t* __restrict__ wt_sa_wq, const bf16_t* __restrict__ wt_sa_wk,
    const bf16_t* __restrict__ wt_sa_wv, const bf16_t* __restrict__ wt_sa_fcw,
    const float* __restrict__ sa_fcb, const float* __restrict__ sa_lng,
    const float* __restrict__ sa_lnb, const bf16_t* __restrict__ wt_ms_w1,
    const float* __restrict__ ms_b1, const bf16_t* __restrict__ wt_ms_w2,
    const float* __restrict__ ms_b2, const float* __restrict__ ca_wq,
    const bf16_t* __restrict__ wt_ca_wk, const bf16_t* __restrict__ wt_ca_wv,
    const float* __restrict__ ca_fcw, const float* __restrict__ ca_fcb,
    const float* __restrict__ ca_lng, const float* __restrict__ ca_lnb,
    const float* __restrict__ mg_w1, const float* __restrict__ mg_b1,
    const float* __restrict__ mg_w2, const float* __restrict__ mg_b2,
    float* __restrict__ out) {
  extern __shared__ char smem[];
  bf16_t* kvb = (bf16_t*)(smem + OFF_KVB);
  bf16_t* kb  = (bf16_t*)(smem + OFF_KB);
  bf16_t* vtb = (bf16_t*)(smem + OFF_VTB);
  bf16_t* aob = (bf16_t*)(smem + OFF_AOB);
  bf16_t* scr = (bf16_t*)(smem + OFF_SCR);
  float*  msk = (float*)(smem + OFF_MSK);
  float*  fb  = (float*)(smem + OFF_F);
  float* qvec = fb;            // 256
  float* qha  = fb + 256;      // 256
  float* sbuf = fb + 512;      // 256
  float* abuf = fb + 768;      // 256
  float* ybuf = fb + 1024;     // 256
  float* conb = fb + 1280;     // 256
  float* hbuf = fb + 1536;     // 128
  float* stat = fb + 1664;     // 8

  const int b    = blockIdx.x;
  const int tid  = threadIdx.x;
  const int lane = tid & 31;
  const int wid  = tid >> 5;
  const int q0   = wid * 16;               // this wave's 16-row block
  bf16_t* scw = scr + wid * 16 * 128;      // per-wave staging 16x128

  // ---- Phase 0: kv = concat(seq, seq_t) -> LDS bf16; additive mask --------
  for (int i = tid; i < Nn * Dd; i += 256) {
    int r = i >> 8, c = i & 255;
    float v = (c < Ff) ? seq[(size_t)b * Nn * Ff + r * Ff + c]
                       : seq_t[(size_t)b * Nn * Ff + r * Ff + (c - Ff)];
    kvb[i] = (bf16_t)v;
  }
  if (tid < Nn) msk[tid] = mask[b * Nn + tid] ? -1e10f : 0.0f;
  __syncthreads();

  // ---- Phase 1: K_sa = kv@wk -> kb ; V_sa^T = (kv@wv)^T -> vtb ------------
  {
    v16bf afr[8];
#pragma unroll
    for (int kt = 0; kt < 8; ++kt) afr[kt] = load_frag(kvb, Dd, q0, kt * 32, lane);
    for (int nt = 0; nt < 16; ++nt) {
      v8f ck = vzero(), cv = vzero();
#pragma unroll
      for (int kt = 0; kt < 8; ++kt) {
        ck = wmma_bf(afr[kt], load_frag(wt_sa_wk, Dd, nt * 16, kt * 32, lane), ck);
        cv = wmma_bf(afr[kt], load_frag(wt_sa_wv, Dd, nt * 16, kt * 32, lane), cv);
      }
      store_c(kb, Dd, q0, nt * 16, ck, lane);
      store_cT(vtb, Nn, q0, nt * 16, cv, lane);
    }
  }
  __syncthreads();

  // ---- Phase 2: per-head self-attention, wave-local softmax ---------------
  for (int h = 0; h < 2; ++h) {
    {  // Q block (16 x 128) for this head -> per-wave staging
      v16bf afr[8];
#pragma unroll
      for (int kt = 0; kt < 8; ++kt) afr[kt] = load_frag(kvb, Dd, q0, kt * 32, lane);
      for (int nt = 0; nt < 8; ++nt) {
        v8f c = vzero();
#pragma unroll
        for (int kt = 0; kt < 8; ++kt)
          c = wmma_bf(afr[kt],
                      load_frag(wt_sa_wq, Dd, h * 128 + nt * 16, kt * 32, lane), c);
        store_c(scw, 128, 0, nt * 16, c, lane);
      }
    }
    // Scores S = Qblk (16x128) @ Kh^T : 8 k-tiles in registers
    v8f s[8];
#pragma unroll
    for (int t = 0; t < 8; ++t) s[t] = vzero();
    for (int dt = 0; dt < 4; ++dt) {
      v16bf a = load_frag(scw, 128, 0, dt * 32, lane);
      for (int t = 0; t < 8; ++t)
        s[t] = wmma_bf(a, load_frag(kb, Dd, t * 16, h * 128 + dt * 32, lane), s[t]);
    }
    int n = lane & 15, hh2 = lane >> 4;
#pragma unroll
    for (int t = 0; t < 8; ++t) {
      float mk = msk[t * 16 + n];
#pragma unroll
      for (int i = 0; i < 8; ++i) s[t][i] = s[t][i] * INV_SCALE + mk;
    }
    // Softmax per row (row = i + 8*half): reduce over 8 tiles + 16 lanes
#pragma unroll
    for (int i = 0; i < 8; ++i) {
      float m = -3.4e38f;
#pragma unroll
      for (int t = 0; t < 8; ++t) m = fmaxf(m, s[t][i]);
      for (int off = 1; off < 16; off <<= 1) m = fmaxf(m, __shfl_xor(m, off, 16));
      float sum = 0.0f;
#pragma unroll
      for (int t = 0; t < 8; ++t) {
        float p = __expf(s[t][i] - m);
        s[t][i] = p;
        sum += p;
      }
      for (int off = 1; off < 16; off <<= 1) sum += __shfl_xor(sum, off, 16);
      float inv = 1.0f / sum;
#pragma unroll
      for (int t = 0; t < 8; ++t) s[t][i] *= inv;
    }
    // attn -> staging (bf16), then O = attn @ Vh via V^T
#pragma unroll
    for (int t = 0; t < 8; ++t)
#pragma unroll
      for (int i = 0; i < 8; ++i)
        scw[(i + 8 * hh2) * 128 + t * 16 + n] = (bf16_t)s[t][i];
    for (int nt = 0; nt < 8; ++nt) {
      v8f c = vzero();
#pragma unroll
      for (int kt = 0; kt < 4; ++kt) {
        v16bf a = load_frag(scw, 128, 0, kt * 32, lane);
        c = wmma_bf(a, load_frag(vtb, Nn, h * 128 + nt * 16, kt * 32, lane), c);
      }
      store_c(aob, Dd, q0, h * 128 + nt * 16, c, lane);
    }
  }
  __syncthreads();

  // ---- Phase 3: sa_out = LN(aob@fcw + fcb + kv) -> kb ---------------------
  {
    v16bf afr[8];
#pragma unroll
    for (int kt = 0; kt < 8; ++kt) afr[kt] = load_frag(aob, Dd, q0, kt * 32, lane);
    v8f acc[16];
    for (int nt = 0; nt < 16; ++nt) {
      acc[nt] = vzero();
#pragma unroll
      for (int kt = 0; kt < 8; ++kt)
        acc[nt] = wmma_bf(afr[kt],
                          load_frag(wt_sa_fcw, Dd, nt * 16, kt * 32, lane), acc[nt]);
    }
    int n = lane & 15, hh2 = lane >> 4;
    float s1[8], s2[8];
#pragma unroll
    for (int i = 0; i < 8; ++i) { s1[i] = 0.0f; s2[i] = 0.0f; }
    for (int nt = 0; nt < 16; ++nt) {
      int col = nt * 16 + n;
      float bias = sa_fcb[col];
#pragma unroll
      for (int i = 0; i < 8; ++i) {
        int row = q0 + i + 8 * hh2;
        float v = acc[nt][i] + bias + (float)kvb[(size_t)row * Dd + col];
        acc[nt][i] = v;
        s1[i] += v;
        s2[i] += v * v;
      }
    }
#pragma unroll
    for (int i = 0; i < 8; ++i)
      for (int off = 1; off < 16; off <<= 1) {
        s1[i] += __shfl_xor(s1[i], off, 16);
        s2[i] += __shfl_xor(s2[i], off, 16);
      }
    for (int nt = 0; nt < 16; ++nt) {
      int col = nt * 16 + n;
      float g = sa_lng[col], be = sa_lnb[col];
#pragma unroll
      for (int i = 0; i < 8; ++i) {
        float mean = s1[i] * (1.0f / 256.0f);
        float var  = s2[i] * (1.0f / 256.0f) - mean * mean;
        float rstd = rsqrtf(var + 1e-5f);
        int row = q0 + i + 8 * hh2;
        kb[(size_t)row * Dd + col] = (bf16_t)((acc[nt][i] - mean) * rstd * g + be);
      }
    }
  }
  __syncthreads();

  // ---- Phase 4: h1 = relu(concat(sa_out, seq)@ms_w1 + b1) -> aob ----------
  {
    v16bf afr[12];
#pragma unroll
    for (int kt = 0; kt < 8; ++kt)  afr[kt] = load_frag(kb, Dd, q0, kt * 32, lane);
#pragma unroll
    for (int kt = 8; kt < 12; ++kt) afr[kt] = load_frag(kvb, Dd, q0, (kt - 8) * 32, lane);
    int n = lane & 15;
    for (int nt = 0; nt < 16; ++nt) {
      v8f c = vzero();
#pragma unroll
      for (int kt = 0; kt < 12; ++kt)
        c = wmma_bf(afr[kt], load_frag(wt_ms_w1, 384, nt * 16, kt * 32, lane), c);
      float bias = ms_b1[nt * 16 + n];
#pragma unroll
      for (int i = 0; i < 8; ++i) c[i] = fmaxf(c[i] + bias, 0.0f);
      store_c(aob, Dd, q0, nt * 16, c, lane);
    }
  }
  __syncthreads();

  // ---- Phase 5: ca_k = h1@ms_w2 + b2 -> kb --------------------------------
  {
    v16bf afr[8];
#pragma unroll
    for (int kt = 0; kt < 8; ++kt) afr[kt] = load_frag(aob, Dd, q0, kt * 32, lane);
    int n = lane & 15;
    for (int nt = 0; nt < 16; ++nt) {
      v8f c = vzero();
#pragma unroll
      for (int kt = 0; kt < 8; ++kt)
        c = wmma_bf(afr[kt], load_frag(wt_ms_w2, Dd, nt * 16, kt * 32, lane), c);
      float bias = ms_b2[nt * 16 + n];
#pragma unroll
      for (int i = 0; i < 8; ++i) c[i] += bias;
      store_c(kb, Dd, q0, nt * 16, c, lane);
    }
  }
  __syncthreads();

  // ---- Phase 6: K_ca = ca_k@ca_wk -> aob ; V_ca = ca_k@ca_wv -> vtb(row) --
  {
    bf16_t* vhb = vtb;  // reinterpret as [128][256] row-major
    v16bf afr[8];
#pragma unroll
    for (int kt = 0; kt < 8; ++kt) afr[kt] = load_frag(kb, Dd, q0, kt * 32, lane);
    for (int nt = 0; nt < 16; ++nt) {
      v8f ck = vzero(), cv = vzero();
#pragma unroll
      for (int kt = 0; kt < 8; ++kt) {
        ck = wmma_bf(afr[kt], load_frag(wt_ca_wk, Dd, nt * 16, kt * 32, lane), ck);
        cv = wmma_bf(afr[kt], load_frag(wt_ca_wv, Dd, nt * 16, kt * 32, lane), cv);
      }
      store_c(aob, Dd, q0, nt * 16, ck, lane);
      store_c(vhb, Dd, q0, nt * 16, cv, lane);
    }
  }
  __syncthreads();

  // ---- Phase 7: 1-query cross-attention tail + output MLP (scalar f32) ----
  if (tid < Dd)
    qvec[tid] = (tid < Ff) ? src[(size_t)b * Ff + tid]
                           : src_t[(size_t)b * Ff + (tid - Ff)];
  __syncthreads();
  if (tid < Dd) {  // qh = q @ ca_wq (f32 weights from L2)
    float acc = 0.0f;
    for (int k = 0; k < Dd; ++k) acc += qvec[k] * ca_wq[(size_t)k * Dd + tid];
    qha[tid] = acc;
  }
  __syncthreads();
  {  // scores: thread = (head, k)
    int hh = tid >> 7, k = tid & 127;
    const bf16_t* kh = aob + (size_t)k * Dd + hh * 128;
    const float* qq = qha + hh * 128;
    float acc = 0.0f;
    for (int d2 = 0; d2 < 128; ++d2) acc += qq[d2] * (float)kh[d2];
    sbuf[tid] = acc * INV_SCALE + msk[k];
  }
  __syncthreads();
  if (wid < 2) {  // softmax per head; also emit attn output rows
    int hh = wid;
    float m = -3.4e38f;
    for (int j = lane; j < 128; j += 32) m = fmaxf(m, sbuf[hh * 128 + j]);
    for (int off = 1; off < 32; off <<= 1) m = fmaxf(m, __shfl_xor(m, off, 32));
    float sum = 0.0f;
    for (int j = lane; j < 128; j += 32) sum += __expf(sbuf[hh * 128 + j] - m);
    for (int off = 1; off < 32; off <<= 1) sum += __shfl_xor(sum, off, 32);
    float inv = 1.0f / sum;
    for (int j = lane; j < 128; j += 32) {
      float p = __expf(sbuf[hh * 128 + j] - m) * inv;
      abuf[hh * 128 + j] = p;
      out[(size_t)Bn * Ff + ((size_t)hh * Bn + b) * Nn + j] = p;
    }
  }
  __syncthreads();
  if (tid < Dd) {  // attn-out row: sum_k attn[h][k] * V_ca[k][d]
    int hh = tid >> 7;
    const bf16_t* vh = (const bf16_t*)(smem + OFF_VTB);
    float acc = 0.0f;
    for (int k = 0; k < 128; ++k)
      acc += abuf[hh * 128 + k] * (float)vh[(size_t)k * Dd + tid];
    ybuf[tid] = acc;
  }
  __syncthreads();
  if (tid < Dd) {  // fc + bias + residual(q)
    float acc = ca_fcb[tid] + qvec[tid];
    for (int e = 0; e < Dd; ++e) acc += ybuf[e] * ca_fcw[(size_t)e * Dd + tid];
    conb[tid] = acc;
  }
  __syncthreads();
  if (wid == 0) {  // LN stats over 256
    float s1 = 0.0f, s2 = 0.0f;
    for (int j = lane; j < Dd; j += 32) { float v = conb[j]; s1 += v; s2 += v * v; }
    for (int off = 1; off < 32; off <<= 1) {
      s1 += __shfl_xor(s1, off, 32);
      s2 += __shfl_xor(s2, off, 32);
    }
    if (lane == 0) {
      float mean = s1 * (1.0f / 256.0f);
      float var  = s2 * (1.0f / 256.0f) - mean * mean;
      stat[0] = mean;
      stat[1] = rsqrtf(var + 1e-5f);
    }
  }
  __syncthreads();
  if (tid < Dd)
    conb[tid] = (conb[tid] - stat[0]) * stat[1] * ca_lng[tid] + ca_lnb[tid];
  __syncthreads();
  if (tid < Ff) {  // mg layer 1: x = concat(ca_out, src)
    float acc = mg_b1[tid];
    for (int e = 0; e < Dd; ++e) acc += conb[e] * mg_w1[(size_t)e * Ff + tid];
    for (int e = 0; e < Ff; ++e)
      acc += src[(size_t)b * Ff + e] * mg_w1[(size_t)(Dd + e) * Ff + tid];
    hbuf[tid] = fmaxf(acc, 0.0f);
  }
  __syncthreads();
  if (tid < Ff) {  // mg layer 2 -> final out
    float acc = mg_b2[tid];
    for (int e = 0; e < Ff; ++e) acc += hbuf[e] * mg_w2[(size_t)e * Ff + tid];
    out[(size_t)b * Ff + tid] = acc;
  }
}

// ---------------------------------------------------------------------------
extern "C" void kernel_launch(void* const* d_in, const int* in_sizes, int n_in,
                              void* d_out, int out_size, void* d_ws,
                              size_t ws_size, hipStream_t stream) {
  (void)in_sizes; (void)n_in; (void)out_size; (void)ws_size;
  const float* src    = (const float*)d_in[0];
  const float* src_t  = (const float*)d_in[1];
  const float* seq    = (const float*)d_in[2];
  const float* seq_t  = (const float*)d_in[3];
  const unsigned char* mask = (const unsigned char*)d_in[4];
  const float* sa_wq  = (const float*)d_in[5];
  const float* sa_wk  = (const float*)d_in[6];
  const float* sa_wv  = (const float*)d_in[7];
  const float* sa_fcw = (const float*)d_in[8];
  const float* sa_fcb = (const float*)d_in[9];
  const float* ms_w1  = (const float*)d_in[10];
  const float* ms_b1  = (const float*)d_in[11];
  const float* ms_w2  = (const float*)d_in[12];
  const float* ms_b2  = (const float*)d_in[13];
  const float* ca_wq  = (const float*)d_in[14];
  const float* ca_wk  = (const float*)d_in[15];
  const float* ca_wv  = (const float*)d_in[16];
  const float* ca_fcw = (const float*)d_in[17];
  const float* ca_fcb = (const float*)d_in[18];
  const float* mg_w1  = (const float*)d_in[19];
  const float* mg_b1  = (const float*)d_in[20];
  const float* mg_w2  = (const float*)d_in[21];
  const float* mg_b2  = (const float*)d_in[22];
  const float* sa_lng = (const float*)d_in[23];
  const float* sa_lnb = (const float*)d_in[24];
  const float* ca_lng = (const float*)d_in[25];
  const float* ca_lnb = (const float*)d_in[26];
  float* out = (float*)d_out;

  // d_ws: pre-transposed bf16 weights (Wt[N][K]); ~1.1 MB total.
  bf16_t* ws = (bf16_t*)d_ws;
  bf16_t* wt_sa_wq  = ws;
  bf16_t* wt_sa_wk  = ws + 65536;
  bf16_t* wt_sa_wv  = ws + 131072;
  bf16_t* wt_sa_fcw = ws + 196608;
  bf16_t* wt_ms_w1  = ws + 262144;   // N=256, K=384
  bf16_t* wt_ms_w2  = ws + 360448;
  bf16_t* wt_ca_wk  = ws + 425984;
  bf16_t* wt_ca_wv  = ws + 491520;

  auto prep = [&](const float* w, bf16_t* wt, int K, int N) {
    int total = K * N;
    wprep<<<(total + 255) / 256, 256, 0, stream>>>(w, wt, K, N);
  };
  prep(sa_wq,  wt_sa_wq,  256, 256);
  prep(sa_wk,  wt_sa_wk,  256, 256);
  prep(sa_wv,  wt_sa_wv,  256, 256);
  prep(sa_fcw, wt_sa_fcw, 256, 256);
  prep(ms_w1,  wt_ms_w1,  384, 256);
  prep(ms_w2,  wt_ms_w2,  256, 256);
  prep(ca_wk,  wt_ca_wk,  256, 256);
  prep(ca_wv,  wt_ca_wv,  256, 256);

  hipFuncSetAttribute(reinterpret_cast<const void*>(&transformer_fused),
                      hipFuncAttributeMaxDynamicSharedMemorySize, SMEM_BYTES);
  transformer_fused<<<Bn, 256, SMEM_BYTES, stream>>>(
      src, src_t, seq, seq_t, mask, wt_sa_wq, wt_sa_wk, wt_sa_wv, wt_sa_fcw,
      sa_fcb, sa_lng, sa_lnb, wt_ms_w1, ms_b1, wt_ms_w2, ms_b2, ca_wq,
      wt_ca_wk, wt_ca_wv, ca_fcw, ca_fcb, ca_lng, ca_lnb, mg_w1, mg_b1, mg_w2,
      mg_b2, out);
}
